// EdgeMLPMixerPolicy_14482629722500
// MI455X (gfx1250) — matrix operand order repaired and checked by reading, not compiled, not run on hardware
//

#include <hip/hip_runtime.h>
#include <hip/hip_bf16.h>
#include <math.h>

// ---------------------------------------------------------------------------
// EdgeMLPMixerPolicy forward, CDNA5 (gfx1250) WMMA implementation.
//   E=250000 edges, B=128 graphs, H=256.
// Dense math via v_wmma_f32_16x16x32_f16 (f32 accumulate). Weights repacked
// fp32->fp16 in WMMA-B-fragment order (one 32B contiguous load per lane).
// All global traffic through explicit addrspace(1) pointers (global_load_*),
// LDS through addrspace(3) (clean ds_load_b128 A-fragments).
// ---------------------------------------------------------------------------

#define E_EDGES 250000
#define B_GRAPH 128
#define H_DIM   256
#define LN_EPS  1e-5f

typedef __attribute__((ext_vector_type(16))) _Float16 v16h;
typedef __attribute__((ext_vector_type(8)))  _Float16 v8h;
typedef __attribute__((ext_vector_type(4)))  _Float16 v4h;
typedef __attribute__((ext_vector_type(8)))  float    v8f;
typedef __attribute__((ext_vector_type(4)))  float    v4f;

#define AS1 __attribute__((address_space(1)))
#define AS3 __attribute__((address_space(3)))
typedef AS1 const float*          gcf;
typedef AS1 float*                gf;
typedef AS1 const v4f*            gcf4;
typedef AS1 const int*            gci;
typedef AS1 const unsigned char*  gcu8;
typedef AS1 const _Float16*       gch;
typedef AS1 _Float16*             gh;
typedef AS1 const v4h*            gcv4h;
typedef AS1 v8h*                  gv8h;
typedef AS1 const v16h*           gcv16;
typedef AS3 const _Float16*       lch;
typedef AS3 _Float16*             lh_t;
typedef AS3 const v8h*            lcv8h;

__device__ inline float gelu_f(float x) {
    return 0.5f * x * (1.0f + erff(x * 0.70710678118654752f));
}

__device__ inline float rsum8(float v) {
    v += __shfl_xor(v, 1, 8);
    v += __shfl_xor(v, 2, 8);
    v += __shfl_xor(v, 4, 8);
    return v;
}

// ---------------------------------------------------------------------------
// 32(M) x 256(N) output tile GEMM, K = 256 or 512, 256 threads (8 waves).
//   wave -> (row-block mrb in {0,1}, col-quarter ch in {0..3}); 4 accum frags.
// sXg: LN'd fp16 activations in LDS, stride sxs halves (multiple of 8).
// Wpg: packed fp16 weights; fragment (kb,nb) at Wp[(kb*16+nb)*32 + lane].
// Epilogue: sOut = fp16( gelu(acc + bias[col]) ).
// ---------------------------------------------------------------------------
__device__ inline void tile_gemm_gelu(const _Float16* sXg, int sxs,
                                      _Float16* sOutg, int sos,
                                      const v16h* Wpg, const float* biasg, int K)
{
    lch   sX   = (lch)sXg;
    lh_t  sOut = (lh_t)sOutg;
    gcv16 Wp   = (gcv16)Wpg;
    gcf   bias = (gcf)biasg;

    const int tid  = threadIdx.x;
    const int wave = tid >> 5;
    const int lane = tid & 31;
    const int mrb  = wave & 1;       // row block (16 rows)
    const int ch   = wave >> 1;      // col quarter (64 cols)
    const int lrow = mrb * 16 + (lane & 15);
    const int hsel = lane >> 4;

    v8f acc[4];
#pragma unroll
    for (int i = 0; i < 4; ++i)
#pragma unroll
        for (int r = 0; r < 8; ++r) acc[i][r] = 0.0f;

    const int KB = K >> 5;
#pragma unroll 2
    for (int kb = 0; kb < KB; ++kb) {
        // A fragment: 16x32 fp16 per CDNA5 lane layout (2 x ds_load_b128).
        lch rp = sX + lrow * sxs + kb * 32 + hsel * 8;
        v8h lo = *(lcv8h)(rp);
        v8h hi = *(lcv8h)(rp + 16);
        v16h a;
#pragma unroll
        for (int i = 0; i < 8; ++i) { a[i] = lo[i]; a[8 + i] = hi[i]; }
#pragma unroll
        for (int nbl = 0; nbl < 4; ++nbl) {
            const int nb = ch * 4 + nbl;
            v16h bfrag = Wp[(size_t)(kb * 16 + nb) * 32 + lane];
            acc[nbl] = __builtin_amdgcn_wmma_f32_16x16x32_f16(
                false, a, false, bfrag, (short)0, acc[nbl], false, false);
        }
    }

    // D layout: VGPR r -> row r (lanes 0-15) / row r+8 (lanes 16-31).
    const int rbase = mrb * 16 + hsel * 8;
#pragma unroll
    for (int nbl = 0; nbl < 4; ++nbl) {
        const int col = (ch * 4 + nbl) * 16 + (lane & 15);
        const float bb = bias[col];
#pragma unroll
        for (int r = 0; r < 8; ++r) {
            float v = gelu_f(acc[nbl][r] + bb);
            sOut[(rbase + r) * sos + col] = (_Float16)v;
        }
    }
}

// ---------------------------------------------------------------------------
// Weight packing: fp32 [K,N] row-major -> fp16 WMMA-B-fragment order.
// ---------------------------------------------------------------------------
__global__ __launch_bounds__(256) void k_pack(const float* __restrict__ Wg,
                                              _Float16* __restrict__ outg,
                                              int K, int N)
{
    gcf W  = (gcf)Wg;
    gh out = (gh)outg;
    int idx = blockIdx.x * 256 + threadIdx.x;
    if (idx >= K * N) return;
    int k = idx / N, n = idx - k * N;
    int kb = k >> 5, kr = k & 31;
    int nb = n >> 4, nr = n & 15;
    int lane = ((kr >> 3) & 1) ? (16 + nr) : nr;
    int i = (kr & 7) + ((kr >> 4) << 3);
    out[(((size_t)kb * (N >> 4) + nb) * 32 + lane) * 16 + i] = (_Float16)W[idx];
}

__global__ __launch_bounds__(256) void k_zero(float* selsumg, float* cntg)
{
    gf selsum = (gf)selsumg;
    gf cnt    = (gf)cntg;
    int idx = blockIdx.x * 256 + threadIdx.x;
    if (idx < B_GRAPH * H_DIM) selsum[idx] = 0.0f;
    if (idx < B_GRAPH) cnt[idx] = 0.0f;
}

// ---------------------------------------------------------------------------
// q_film: qf = gelu(LN(question) @ qf_W + qf_bias), 32 graphs per block.
// ---------------------------------------------------------------------------
__global__ __launch_bounds__(256) void k_qfilm(const float* __restrict__ qg,
                                               const float* __restrict__ gg,
                                               const float* __restrict__ bg_,
                                               const v16h* __restrict__ Wp,
                                               const float* __restrict__ bias,
                                               _Float16* __restrict__ qf16g)
{
    __shared__ _Float16 sX[32][272];
    __shared__ _Float16 sO[32][272];
    gcf g = (gcf)gg; gcf b = (gcf)bg_;
    const int tid = threadIdx.x, row = tid >> 3, sub = tid & 7;
    const int gr = blockIdx.x * 32 + row;

    float vals[32];
    float s = 0.0f;
    {
        gcf4 pq = (gcf4)((const float*)qg + (size_t)gr * H_DIM + sub * 32);
#pragma unroll
        for (int i = 0; i < 8; ++i) {
            v4f a = pq[i];
#pragma unroll
            for (int k = 0; k < 4; ++k) { vals[i * 4 + k] = a[k]; s += a[k]; }
        }
    }
    float mean = rsum8(s) * (1.0f / H_DIM);
    float sq = 0.0f;
#pragma unroll
    for (int j = 0; j < 32; ++j) { float d = vals[j] - mean; sq += d * d; }
    float rstd = rsqrtf(rsum8(sq) * (1.0f / H_DIM) + LN_EPS);
#pragma unroll
    for (int j = 0; j < 32; ++j) {
        int c = sub * 32 + j;
        sX[row][c] = (_Float16)((vals[j] - mean) * rstd * g[c] + b[c]);
    }
    __syncthreads();
    tile_gemm_gelu(&sX[0][0], 272, &sO[0][0], 272, Wp, bias, 256);
    __syncthreads();
    {
        gv8h pout = (gv8h)((_Float16*)qf16g + (size_t)gr * H_DIM + sub * 32);
#pragma unroll
        for (int i = 0; i < 4; ++i) {
            v8h hv;
#pragma unroll
            for (int k = 0; k < 8; ++k) hv[k] = sO[row][sub * 32 + i * 8 + k];
            pout[i] = hv;
        }
    }
}

// ---------------------------------------------------------------------------
// Phase 1: embeddings + FiLM gather -> LN -> GEMM1 -> GELU -> LN -> GEMM2
//          -> GELU -> store h (fp16) + masked atomic segment-sum.
// ---------------------------------------------------------------------------
__global__ __launch_bounds__(256) void k_phase1(
    const float* __restrict__ edge_tokens, const int* __restrict__ ebg,
    const unsigned char* __restrict__ maskg, const int* __restrict__ sog,
    const int* __restrict__ edg,
    const float* __restrict__ type_emb, const float* __restrict__ order_emb,
    const float* __restrict__ dir_emb, const _Float16* __restrict__ qf16,
    const float* __restrict__ g1g, const float* __restrict__ b1g,
    const v16h* __restrict__ W1p, const float* __restrict__ bb1,
    const float* __restrict__ g2g, const float* __restrict__ b2g,
    const v16h* __restrict__ W2p, const float* __restrict__ bb2,
    _Float16* __restrict__ hbuf, float* __restrict__ selsum,
    float* __restrict__ cnt)
{
    __shared__ _Float16 sX[32][272];
    __shared__ _Float16 sH[32][272];
    gcf g1 = (gcf)g1g; gcf b1 = (gcf)b1g;
    gcf g2 = (gcf)g2g; gcf b2 = (gcf)b2g;
    const int tid = threadIdx.x, row = tid >> 3, sub = tid & 7;
    const int e = blockIdx.x * 32 + row;
    const bool valid = e < E_EDGES;

    int bg = 0, oi = 0, di = 0; bool msk = false;
    if (valid) {
        bg = ((gci)ebg)[e];
        int so = ((gci)sog)[e]; so = so < -1 ? -1 : (so > 10 ? 10 : so); oi = so + 1;
        int d  = ((gci)edg)[e]; di = d < 0 ? 0 : (d > 2 ? 2 : d);
        msk = ((gcu8)maskg)[e] != 0;
    }

    float vals[32];
    float s = 0.0f;
    if (valid) {
        gcf4  pe = (gcf4)(edge_tokens + (size_t)e * H_DIM + sub * 32);
        gcf4  po = (gcf4)(order_emb + (size_t)oi * H_DIM + sub * 32);
        gcf4  pd = (gcf4)(dir_emb + (size_t)di * H_DIM + sub * 32);
        gcf4  pt = (gcf4)(type_emb + 2 * H_DIM + sub * 32);
        gcv4h pq = (gcv4h)(qf16 + (size_t)bg * H_DIM + sub * 32);
#pragma unroll
        for (int i = 0; i < 8; ++i) {
            v4f a = pe[i] + po[i] + pd[i] + pt[i];
            v4h q = pq[i];
#pragma unroll
            for (int k = 0; k < 4; ++k) {
                float v = a[k] + (float)q[k];
                vals[i * 4 + k] = v; s += v;
            }
        }
    } else {
#pragma unroll
        for (int j = 0; j < 32; ++j) vals[j] = 0.0f;
    }
    float mean = rsum8(s) * (1.0f / H_DIM);
    float sq = 0.0f;
#pragma unroll
    for (int j = 0; j < 32; ++j) { float d = vals[j] - mean; sq += d * d; }
    float rstd = rsqrtf(rsum8(sq) * (1.0f / H_DIM) + LN_EPS);
#pragma unroll
    for (int j = 0; j < 32; ++j) {
        int c = sub * 32 + j;
        sX[row][c] = (_Float16)((vals[j] - mean) * rstd * g1[c] + b1[c]);
    }
    __syncthreads();
    tile_gemm_gelu(&sX[0][0], 272, &sH[0][0], 272, W1p, bb1, 256);
    __syncthreads();

    // second LN over h1
    s = 0.0f;
#pragma unroll
    for (int j = 0; j < 32; ++j) { vals[j] = (float)sH[row][sub * 32 + j]; s += vals[j]; }
    mean = rsum8(s) * (1.0f / H_DIM);
    sq = 0.0f;
#pragma unroll
    for (int j = 0; j < 32; ++j) { float d = vals[j] - mean; sq += d * d; }
    rstd = rsqrtf(rsum8(sq) * (1.0f / H_DIM) + LN_EPS);
    __syncthreads();
#pragma unroll
    for (int j = 0; j < 32; ++j) {
        int c = sub * 32 + j;
        sX[row][c] = (_Float16)((vals[j] - mean) * rstd * g2[c] + b2[c]);
    }
    __syncthreads();
    tile_gemm_gelu(&sX[0][0], 272, &sH[0][0], 272, W2p, bb2, 256);
    __syncthreads();

    if (valid) {
        gv8h pout = (gv8h)(hbuf + (size_t)e * H_DIM + sub * 32);
#pragma unroll
        for (int i = 0; i < 4; ++i) {
            v8h hv;
#pragma unroll
            for (int k = 0; k < 8; ++k) hv[k] = sH[row][sub * 32 + i * 8 + k];
            pout[i] = hv;
            if (msk) {
#pragma unroll
                for (int k = 0; k < 8; ++k)
                    atomicAdd(&selsum[bg * H_DIM + sub * 32 + i * 8 + k], (float)hv[k]);
            }
        }
        if (msk && sub == 0) atomicAdd(&cnt[bg], 1.0f);
    }
}

// ---------------------------------------------------------------------------
// cur = LN(sel_sum/max(cnt,1) + state); stop head on [cur, state] (K=512).
// ---------------------------------------------------------------------------
__global__ __launch_bounds__(256) void k_curstop(
    const float* __restrict__ selsumg, const float* __restrict__ cntg,
    const float* __restrict__ stateg,
    const float* __restrict__ gngg, const float* __restrict__ gnbg,
    const float* __restrict__ spgg, const float* __restrict__ spbg,
    const v16h* __restrict__ spW1p, const float* __restrict__ spb1,
    const float* __restrict__ spW2g, const float* __restrict__ spb2g,
    _Float16* __restrict__ cur16g, float* __restrict__ out_curg,
    float* __restrict__ out_stopg)
{
    __shared__ _Float16 sX[32][528];
    __shared__ _Float16 sO[32][272];
    gcf gng = (gcf)gngg; gcf gnb = (gcf)gnbg;
    gcf spg = (gcf)spgg; gcf spb = (gcf)spbg;
    const int tid = threadIdx.x, row = tid >> 3, sub = tid & 7;
    const int g = blockIdx.x * 32 + row;

    float cn = ((gcf)cntg)[g]; cn = cn < 1.0f ? 1.0f : cn;
    const float inv = 1.0f / cn;

    float vals[32];
    float s = 0.0f;
    {
        gcf4 ps = (gcf4)(selsumg + (size_t)g * H_DIM + sub * 32);
        gcf4 pt = (gcf4)(stateg + (size_t)g * H_DIM + sub * 32);
#pragma unroll
        for (int i = 0; i < 8; ++i) {
            v4f a = ps[i] * inv + pt[i];
#pragma unroll
            for (int k = 0; k < 4; ++k) { vals[i * 4 + k] = a[k]; s += a[k]; }
        }
    }
    float mean = rsum8(s) * (1.0f / H_DIM);
    float sq = 0.0f;
#pragma unroll
    for (int j = 0; j < 32; ++j) { float d = vals[j] - mean; sq += d * d; }
    float rstd = rsqrtf(rsum8(sq) * (1.0f / H_DIM) + LN_EPS);
#pragma unroll
    for (int j = 0; j < 32; ++j) {
        int c = sub * 32 + j;
        float cv = (vals[j] - mean) * rstd * gng[c] + gnb[c];
        ((gf)out_curg)[g * H_DIM + c] = cv;
        ((gh)cur16g)[g * H_DIM + c]   = (_Float16)cv;
        sO[row][c]                    = (_Float16)cv;   // stash for 512-wide LN
    }
    __syncthreads();

    {   // LN over concat [cur, state] (512); sub<4 -> cur part, sub>=4 -> state
        float v2[64];
        float s2 = 0.0f;
        if (sub < 4) {
#pragma unroll
            for (int j = 0; j < 64; ++j) { v2[j] = (float)sO[row][sub * 64 + j]; s2 += v2[j]; }
        } else {
            gcf4 pt = (gcf4)(stateg + (size_t)g * H_DIM + (sub - 4) * 64);
#pragma unroll
            for (int i = 0; i < 16; ++i) {
                v4f a = pt[i];
#pragma unroll
                for (int k = 0; k < 4; ++k) { v2[i * 4 + k] = a[k]; s2 += a[k]; }
            }
        }
        float m2 = rsum8(s2) * (1.0f / 512.0f);
        float q2 = 0.0f;
#pragma unroll
        for (int j = 0; j < 64; ++j) { float d = v2[j] - m2; q2 += d * d; }
        float r2 = rsqrtf(rsum8(q2) * (1.0f / 512.0f) + LN_EPS);
#pragma unroll
        for (int j = 0; j < 64; ++j) {
            int c = sub * 64 + j;
            sX[row][c] = (_Float16)((v2[j] - m2) * r2 * spg[c] + spb[c]);
        }
    }
    __syncthreads();
    tile_gemm_gelu(&sX[0][0], 528, &sO[0][0], 272, spW1p, spb1, 512);
    __syncthreads();

    float part = 0.0f;
    {
        gcf w2 = (gcf)spW2g;
#pragma unroll
        for (int j = 0; j < 32; ++j) {
            int c = sub * 32 + j;
            part += (float)sO[row][c] * w2[c];
        }
    }
    part = rsum8(part);
    if (sub == 0) ((gf)out_stopg)[g] = part + ((gcf)spb2g)[0];
}

// ---------------------------------------------------------------------------
// Phase 2: nxt LN -> lh_in = [cur[b], nxt] -> LN(512) -> GEMM(K=512) -> GELU
//          -> dot(lh_W2) -> edge_logits.
// ---------------------------------------------------------------------------
__global__ __launch_bounds__(256) void k_phase2(
    const _Float16* __restrict__ hbuf, const int* __restrict__ ebg,
    const float* __restrict__ selsumg, const float* __restrict__ cntg,
    const float* __restrict__ stateg,
    const float* __restrict__ gngg, const float* __restrict__ gnbg,
    const _Float16* __restrict__ cur16,
    const float* __restrict__ lhgg, const float* __restrict__ lhbg,
    const v16h* __restrict__ lhW1p, const float* __restrict__ lhb1,
    const float* __restrict__ lhW2g, const float* __restrict__ lhb2g,
    float* __restrict__ out_edgeg)
{
    __shared__ _Float16 sX[32][528];
    __shared__ _Float16 sH[32][272];
    gcf gng = (gcf)gngg; gcf gnb = (gcf)gnbg;
    gcf lhg = (gcf)lhgg; gcf lhb = (gcf)lhbg;
    const int tid = threadIdx.x, row = tid >> 3, sub = tid & 7;
    const int e = blockIdx.x * 32 + row;
    const bool valid = e < E_EDGES;
    const int bg = valid ? ((gci)ebg)[e] : 0;
    const float inv = 1.0f / (((gcf)cntg)[bg] + 1.0f);

    float vals[32];
    float s = 0.0f;
    if (valid) {
        gcf4  ps = (gcf4)(selsumg + (size_t)bg * H_DIM + sub * 32);
        gcf4  pt = (gcf4)(stateg + (size_t)bg * H_DIM + sub * 32);
        gcv4h ph = (gcv4h)(hbuf + (size_t)e * H_DIM + sub * 32);
#pragma unroll
        for (int i = 0; i < 8; ++i) {
            v4f a = ps[i];
            v4h hh = ph[i];
            v4f st = pt[i];
#pragma unroll
            for (int k = 0; k < 4; ++k) {
                float v = (a[k] + (float)hh[k]) * inv + st[k];
                vals[i * 4 + k] = v; s += v;
            }
        }
    } else {
#pragma unroll
        for (int j = 0; j < 32; ++j) vals[j] = 0.0f;
    }
    float mean = rsum8(s) * (1.0f / H_DIM);
    float sq = 0.0f;
#pragma unroll
    for (int j = 0; j < 32; ++j) { float d = vals[j] - mean; sq += d * d; }
    float rstd = rsqrtf(rsum8(sq) * (1.0f / H_DIM) + LN_EPS);
#pragma unroll
    for (int j = 0; j < 32; ++j) {
        int c = sub * 32 + j;
        sH[row][c] = (_Float16)((vals[j] - mean) * rstd * gng[c] + gnb[c]); // nxt stash
    }
    __syncthreads();

    {   // LN over lh_in = [cur[b], nxt] (512); sub<4 -> cur, sub>=4 -> nxt
        float v2[64];
        float s2 = 0.0f;
        if (sub < 4) {
            gcv4h pc = (gcv4h)(cur16 + (size_t)bg * H_DIM + sub * 64);
#pragma unroll
            for (int i = 0; i < 16; ++i) {
                v4h q = pc[i];
#pragma unroll
                for (int k = 0; k < 4; ++k) { v2[i * 4 + k] = (float)q[k]; s2 += v2[i * 4 + k]; }
            }
        } else {
#pragma unroll
            for (int j = 0; j < 64; ++j) { v2[j] = (float)sH[row][(sub - 4) * 64 + j]; s2 += v2[j]; }
        }
        float m2 = rsum8(s2) * (1.0f / 512.0f);
        float q2 = 0.0f;
#pragma unroll
        for (int j = 0; j < 64; ++j) { float d = v2[j] - m2; q2 += d * d; }
        float r2 = rsqrtf(rsum8(q2) * (1.0f / 512.0f) + LN_EPS);
        __syncthreads();   // all reads of sH stash complete before GEMM overwrites
#pragma unroll
        for (int j = 0; j < 64; ++j) {
            int c = sub * 64 + j;
            sX[row][c] = (_Float16)((v2[j] - m2) * r2 * lhg[c] + lhb[c]);
        }
    }
    __syncthreads();
    tile_gemm_gelu(&sX[0][0], 528, &sH[0][0], 272, lhW1p, lhb1, 512);
    __syncthreads();

    float part = 0.0f;
    {
        gcf w2 = (gcf)lhW2g;
#pragma unroll
        for (int j = 0; j < 32; ++j) {
            int c = sub * 32 + j;
            part += (float)sH[row][c] * w2[c];
        }
    }
    part = rsum8(part);
    if (sub == 0 && valid) ((gf)out_edgeg)[e] = part + ((gcf)lhb2g)[0];
}

// ---------------------------------------------------------------------------
// Host launcher
// ---------------------------------------------------------------------------
extern "C" void kernel_launch(void* const* d_in, const int* in_sizes, int n_in,
                              void* d_out, int out_size, void* d_ws, size_t ws_size,
                              hipStream_t stream)
{
    (void)in_sizes; (void)n_in; (void)out_size; (void)ws_size;

    const float* edge_tokens     = (const float*)d_in[0];
    const float* question_tokens = (const float*)d_in[1];
    const float* state_tokens    = (const float*)d_in[2];
    const int*   edge_batch      = (const int*)d_in[3];
    const unsigned char* sel_mask = (const unsigned char*)d_in[4]; // jnp bool
    const int*   selection_order = (const int*)d_in[5];
    const int*   edge_direction  = (const int*)d_in[6];
    const float* type_emb  = (const float*)d_in[7];
    const float* order_emb = (const float*)d_in[8];
    const float* dir_emb   = (const float*)d_in[9];
    const float* gn_g = (const float*)d_in[10];
    const float* gn_b = (const float*)d_in[11];
    const float* qf_g = (const float*)d_in[12];
    const float* qf_b = (const float*)d_in[13];
    const float* qf_W = (const float*)d_in[14];
    const float* qf_bias = (const float*)d_in[15];
    const float* m_g1 = (const float*)d_in[16];
    const float* m_b1 = (const float*)d_in[17];
    const float* m_W1 = (const float*)d_in[18];
    const float* m_bb1 = (const float*)d_in[19];
    const float* m_g2 = (const float*)d_in[20];
    const float* m_b2 = (const float*)d_in[21];
    const float* m_W2 = (const float*)d_in[22];
    const float* m_bb2 = (const float*)d_in[23];
    const float* lh_g = (const float*)d_in[24];
    const float* lh_b = (const float*)d_in[25];
    const float* lh_W1 = (const float*)d_in[26];
    const float* lh_b1v = (const float*)d_in[27];
    const float* lh_W2 = (const float*)d_in[28];
    const float* lh_b2v = (const float*)d_in[29];
    const float* sp_g = (const float*)d_in[30];
    const float* sp_b = (const float*)d_in[31];
    const float* sp_W1 = (const float*)d_in[32];
    const float* sp_b1v = (const float*)d_in[33];
    const float* sp_W2 = (const float*)d_in[34];
    const float* sp_b2v = (const float*)d_in[35];

    // outputs: edge_logits [E], stop_logits [B], cur [B,H]
    float* out_edge = (float*)d_out;
    float* out_stop = out_edge + E_EDGES;
    float* out_cur  = out_stop + B_GRAPH;

    // workspace layout
    char* ws = (char*)d_ws;
    _Float16* qfWp  = (_Float16*)(ws + 0);         // 256x256 fp16 packed
    _Float16* mW1p  = (_Float16*)(ws + 131072);
    _Float16* mW2p  = (_Float16*)(ws + 262144);
    _Float16* lhW1p = (_Float16*)(ws + 393216);    // 512x256
    _Float16* spW1p = (_Float16*)(ws + 655360);    // 512x256
    _Float16* qf16  = (_Float16*)(ws + 917504);    // [B,H]
    _Float16* cur16 = (_Float16*)(ws + 983040);    // [B,H]
    float*    selsum = (float*)(ws + 1048576);     // [B,H] f32
    float*    cntf   = (float*)(ws + 1179648);     // [B] f32
    _Float16* hbuf   = (_Float16*)(ws + 1180160);  // [E,H] fp16 (~128 MB)

    // 0) pack weights + zero accumulators
    k_pack<<<(256 * 256 + 255) / 256, 256, 0, stream>>>(qf_W, qfWp, 256, 256);
    k_pack<<<(256 * 256 + 255) / 256, 256, 0, stream>>>(m_W1, mW1p, 256, 256);
    k_pack<<<(256 * 256 + 255) / 256, 256, 0, stream>>>(m_W2, mW2p, 256, 256);
    k_pack<<<(512 * 256 + 255) / 256, 256, 0, stream>>>(lh_W1, lhW1p, 512, 256);
    k_pack<<<(512 * 256 + 255) / 256, 256, 0, stream>>>(sp_W1, spW1p, 512, 256);
    k_zero<<<(B_GRAPH * H_DIM + 255) / 256, 256, 0, stream>>>(selsum, cntf);

    // 1) q_film
    k_qfilm<<<B_GRAPH / 32, 256, 0, stream>>>(question_tokens, qf_g, qf_b,
                                              (const v16h*)qfWp, qf_bias, qf16);

    // 2) edge MLP + segment sums
    const int etiles = (E_EDGES + 31) / 32;
    k_phase1<<<etiles, 256, 0, stream>>>(edge_tokens, edge_batch, sel_mask,
                                         selection_order, edge_direction,
                                         type_emb, order_emb, dir_emb, qf16,
                                         m_g1, m_b1, (const v16h*)mW1p, m_bb1,
                                         m_g2, m_b2, (const v16h*)mW2p, m_bb2,
                                         hbuf, selsum, cntf);

    // 3) cur + stop head
    k_curstop<<<B_GRAPH / 32, 256, 0, stream>>>(selsum, cntf, state_tokens,
                                                gn_g, gn_b, sp_g, sp_b,
                                                (const v16h*)spW1p, sp_b1v,
                                                sp_W2, sp_b2v,
                                                cur16, out_cur, out_stop);

    // 4) lookahead head -> edge logits
    k_phase2<<<etiles, 256, 0, stream>>>(hbuf, edge_batch, selsum, cntf,
                                         state_tokens, gn_g, gn_b, cur16,
                                         lh_g, lh_b, (const v16h*)lhW1p, lh_b1v,
                                         lh_W2, lh_b2v, out_edge);
}
